// aamsoftmax_88845693485822
// MI455X (gfx1250) — compile-verified, matrix operand order checked by reading, not run on hardware
//
#include <hip/hip_runtime.h>

// Problem constants (match the reference)
#define B_ROWS 2048
#define D_DIM  512
#define C_CLS  50000
#define NBLK   391   // ceil(C_CLS / 128) column blocks

__device__ __constant__ const float SCALE_S = 15.0f;
// m = pi/6
__device__ __constant__ const float COS_M = 0.8660254037844387f;   // cos(m)
__device__ __constant__ const float SIN_M = 0.49999999999999994f;  // sin(m)
__device__ __constant__ const float TH_C  = -0.8660254037844387f;  // cos(pi - m)
__device__ __constant__ const float MM_C  = 0.26179938779914946f;  // sin(pi - m) * m

typedef _Float16 v16h __attribute__((ext_vector_type(16)));
typedef _Float16 h8   __attribute__((ext_vector_type(8)));
typedef _Float16 h4   __attribute__((ext_vector_type(4)));
typedef float    v8f  __attribute__((ext_vector_type(8)));

// ---------------------------------------------------------------------------
// L2-normalize rows of a [rows x 512] f32 matrix into f16.
// One wave (32 lanes) per row; each lane handles 4 float4 chunks (16 floats).
// ---------------------------------------------------------------------------
__global__ __launch_bounds__(256) void l2_normalize_f16(const float* __restrict__ in,
                                                        _Float16* __restrict__ out,
                                                        int rows) {
  const int lane = threadIdx.x & 31;
  const int row  = blockIdx.x * 8 + (threadIdx.x >> 5);
  if (row >= rows) return;  // whole wave exits together (row is wave-uniform)

  const float4* p = (const float4*)(in + (size_t)row * D_DIM);
  float4 v[4];
  float ss = 0.0f;
#pragma unroll
  for (int i = 0; i < 4; ++i) {
    v[i] = p[lane + 32 * i];
    ss += v[i].x * v[i].x + v[i].y * v[i].y + v[i].z * v[i].z + v[i].w * v[i].w;
  }
#pragma unroll
  for (int off = 16; off > 0; off >>= 1) ss += __shfl_xor(ss, off, 32);

  const float sc = 1.0f / fmaxf(sqrtf(ss), 1e-12f);

  h4* o = (h4*)(out + (size_t)row * D_DIM);
#pragma unroll
  for (int i = 0; i < 4; ++i) {
    h4 q = { (_Float16)(v[i].x * sc), (_Float16)(v[i].y * sc),
             (_Float16)(v[i].z * sc), (_Float16)(v[i].w * sc) };
    o[lane + 32 * i] = q;
  }
}

// ---------------------------------------------------------------------------
// Fused GEMM + margin + partial-logsumexp.
//   out[B,C] = S * margin(xn @ wn^T)           (nontemporal stores)
//   part_m/part_s[B][NBLK] = per-(row, col-block) online (max, sumexp)
//   tgt[B] = final target-class logit (post-margin, scaled)
//
// Block tile 128(M) x 128(N); 8 waves arranged 4(M) x 2(N); each wave does
// 32x64 = 2x4 tiles of 16x16 via v_wmma_f32_16x16x32_f16. K = 512 -> 16 steps.
//
// Fragment layout per CDNA5 ISA (16-bit A 16x32): lane L (h=L>>4, l=L&15)
// holds row (base+l); its 16 halves cover K = {8h..8h+7} and {16+8h..16+8h+7}
// within each K-32 chunk -> two aligned 16B loads per fragment. B (32x16,
// wn^T) mirrors the layout with l -> column, i.e. the same gather from
// row-major wn.
// ---------------------------------------------------------------------------
__global__ __launch_bounds__(256) void cosine_gemm_wmma(const _Float16* __restrict__ xn,
                                                        const _Float16* __restrict__ wn,
                                                        const int* __restrict__ label,
                                                        float* __restrict__ out,
                                                        float* __restrict__ part_m,
                                                        float* __restrict__ part_s,
                                                        float* __restrict__ tgt) {
  const int lane  = threadIdx.x & 31;
  const int wave  = threadIdx.x >> 5;
  const int h     = lane >> 4;
  const int l     = lane & 15;
  const int waveM = wave >> 1;   // 0..3
  const int waveN = wave & 1;    // 0..1
  const int mWave = blockIdx.y * 128 + waveM * 32;
  const int nWave = blockIdx.x * 128 + waveN * 64;

  __shared__ int   lbl_s[128];
  __shared__ float pm_s[128][2];
  __shared__ float ps_s[128][2];
  if (threadIdx.x < 128) lbl_s[threadIdx.x] = label[blockIdx.y * 128 + threadIdx.x];

  int ra[2];
#pragma unroll
  for (int mt = 0; mt < 2; ++mt) ra[mt] = mWave + mt * 16 + l;  // always < 2048
  int rb[4];
#pragma unroll
  for (int nt = 0; nt < 4; ++nt) {
    int r = nWave + nt * 16 + l;
    rb[nt] = (r < C_CLS) ? r : (C_CLS - 1);  // clamp loads; stores are masked
  }

  v8f acc[2][4] = {};

  for (int ks = 0; ks < D_DIM / 32; ++ks) {
    const int off = ks * 32 + h * 8;
    v16h a[2], b[4];
#pragma unroll
    for (int mt = 0; mt < 2; ++mt) {
      const h8* p = (const h8*)(xn + (size_t)ra[mt] * D_DIM + off);
      a[mt] = __builtin_shufflevector(p[0], p[2], 0, 1, 2, 3, 4, 5, 6, 7,
                                      8, 9, 10, 11, 12, 13, 14, 15);
    }
#pragma unroll
    for (int nt = 0; nt < 4; ++nt) {
      const h8* p = (const h8*)(wn + (size_t)rb[nt] * D_DIM + off);
      b[nt] = __builtin_shufflevector(p[0], p[2], 0, 1, 2, 3, 4, 5, 6, 7,
                                      8, 9, 10, 11, 12, 13, 14, 15);
    }
#pragma unroll
    for (int mt = 0; mt < 2; ++mt)
#pragma unroll
      for (int nt = 0; nt < 4; ++nt)
        acc[mt][nt] = __builtin_amdgcn_wmma_f32_16x16x32_f16(
            /*neg_a=*/false, a[mt], /*neg_b=*/false, b[nt],
            /*c_mod=*/(short)0, acc[mt][nt],
            /*reuse_a=*/false, /*reuse_b=*/false);
  }

  __syncthreads();  // lbl_s ready (also orders the epilogue LDS writes)

  // Epilogue. C/D layout: VGPR v holds (M = tileRow + v + 8h, N = tileCol + l).
  // Per (mt, v): one output row; its 4 nt-values (+15 neighbor lanes of the
  // same half-wave) span 64 contiguous columns.
#pragma unroll
  for (int mt = 0; mt < 2; ++mt) {
#pragma unroll
    for (int v = 0; v < 8; ++v) {
      const int row_local = waveM * 32 + mt * 16 + h * 8 + v;
      const int row_g     = blockIdx.y * 128 + row_local;
      const int lbl       = lbl_s[row_local];
      float m = -__builtin_inff();
      float s = 0.0f;
#pragma unroll
      for (int nt = 0; nt < 4; ++nt) {
        const int col = nWave + nt * 16 + l;
        if (col < C_CLS) {
          const float c = acc[mt][nt][v];  // cosine (unscaled)
          float val;
          if (col == lbl) {
            const float sin_t = sqrtf(fminf(fmaxf(1.0f - c * c, 0.0f), 1.0f));
            float phi = c * COS_M - sin_t * SIN_M;
            phi = (c - TH_C > 0.0f) ? phi : (c - MM_C);
            val = SCALE_S * phi;
            tgt[row_g] = val;  // unique writer per row across the whole grid
          } else {
            val = SCALE_S * c;
          }
          __builtin_nontemporal_store(val, &out[(size_t)row_g * C_CLS + col]);
          const float nm = fmaxf(m, val);
          s = s * __expf(m - nm) + __expf(val - nm);
          m = nm;
        }
      }
      // Merge across the 16 lanes (same h) that share this row.
#pragma unroll
      for (int msk = 1; msk < 16; msk <<= 1) {
        const float m2 = __shfl_xor(m, msk, 32);
        const float s2 = __shfl_xor(s, msk, 32);
        const float nm = fmaxf(m, m2);
        s = s * __expf(m - nm) + s2 * __expf(m2 - nm);
        m = nm;
      }
      if (l == 0) { pm_s[row_local][waveN] = m; ps_s[row_local][waveN] = s; }
    }
  }
  __syncthreads();
  if (threadIdx.x < 128) {
    const float m0 = pm_s[threadIdx.x][0], s0 = ps_s[threadIdx.x][0];
    const float m1 = pm_s[threadIdx.x][1], s1 = ps_s[threadIdx.x][1];
    const float nm = fmaxf(m0, m1);
    const float sc = s0 * __expf(m0 - nm) + s1 * __expf(m1 - nm);
    const int row_g = blockIdx.y * 128 + threadIdx.x;
    part_m[(size_t)row_g * NBLK + blockIdx.x] = nm;
    part_s[(size_t)row_g * NBLK + blockIdx.x] = sc;
  }
}

// ---------------------------------------------------------------------------
// Zero the scalar loss accumulator (d_out[B*C]).
// ---------------------------------------------------------------------------
__global__ void zero_loss_kernel(float* loss_slot) { *loss_slot = 0.0f; }

// ---------------------------------------------------------------------------
// Final loss: per row, merge NBLK (max,sumexp) partials -> logsumexp, then
// loss_i = lse - target_logit; atomically accumulate mean into loss slot.
// ---------------------------------------------------------------------------
__global__ __launch_bounds__(256) void loss_from_partials(const float* __restrict__ part_m,
                                                          const float* __restrict__ part_s,
                                                          const float* __restrict__ tgt,
                                                          float* __restrict__ loss_slot) {
  const int row = blockIdx.x;
  const int tid = threadIdx.x;
  __shared__ float sm[256];
  __shared__ float ss[256];

  float m = -__builtin_inff();
  float s = 0.0f;
  for (int nb = tid; nb < NBLK; nb += 256) {
    const float m2 = part_m[(size_t)row * NBLK + nb];
    const float s2 = part_s[(size_t)row * NBLK + nb];
    const float nm = fmaxf(m, m2);
    s = s * __expf(m - nm) + s2 * __expf(m2 - nm);
    m = nm;
  }
  sm[tid] = m;
  ss[tid] = s;
  __syncthreads();
  for (int off = 128; off > 0; off >>= 1) {
    if (tid < off) {
      const float m2 = sm[tid + off], s2 = ss[tid + off];
      const float nm = fmaxf(sm[tid], m2);
      ss[tid] = ss[tid] * __expf(sm[tid] - nm) + s2 * __expf(m2 - nm);
      sm[tid] = nm;
    }
    __syncthreads();
  }
  if (tid == 0) {
    const float lse = sm[0] + __logf(ss[0]);
    atomicAdd(loss_slot, (lse - tgt[row]) * (1.0f / (float)B_ROWS));
  }
}

// ---------------------------------------------------------------------------
// Launch
// ---------------------------------------------------------------------------
extern "C" void kernel_launch(void* const* d_in, const int* in_sizes, int n_in,
                              void* d_out, int out_size, void* d_ws, size_t ws_size,
                              hipStream_t stream) {
  const float* x      = (const float*)d_in[0];
  const int*   label  = (const int*)d_in[1];
  const float* weight = (const float*)d_in[2];
  float* out = (float*)d_out;

  // Workspace layout:
  //   xn  f16 [2048 x 512]      (2 MB)
  //   wn  f16 [50000 x 512]     (51.2 MB)
  //   part_m f32 [2048 x 391]   (3.2 MB)
  //   part_s f32 [2048 x 391]   (3.2 MB)
  //   tgt    f32 [2048]         (8 KB)
  _Float16* xn = (_Float16*)d_ws;
  _Float16* wn = xn + (size_t)B_ROWS * D_DIM;
  float* part_m = (float*)(wn + (size_t)C_CLS * D_DIM);
  float* part_s = part_m + (size_t)B_ROWS * NBLK;
  float* tgt    = part_s + (size_t)B_ROWS * NBLK;
  float* loss_slot = out + (size_t)B_ROWS * C_CLS;

  l2_normalize_f16<<<(B_ROWS + 7) / 8, 256, 0, stream>>>(x, xn, B_ROWS);
  l2_normalize_f16<<<(C_CLS + 7) / 8, 256, 0, stream>>>(weight, wn, C_CLS);
  zero_loss_kernel<<<1, 1, 0, stream>>>(loss_slot);

  dim3 grid(NBLK, B_ROWS / 128);  // (391, 16)
  cosine_gemm_wmma<<<grid, 256, 0, stream>>>(xn, wn, label, out, part_m, part_s, tgt);

  loss_from_partials<<<B_ROWS, 256, 0, stream>>>(part_m, part_s, tgt, loss_slot);
}
